// Attention_72722386255902
// MI455X (gfx1250) — matrix operand order repaired and checked by reading, not compiled, not run on hardware
//
#include <hip/hip_runtime.h>
#include <cstdint>
#include <cstddef>

// ---------------------------------------------------------------------------
// CDNA5 (gfx1250) fused attention:  wave32, WMMA bf16 16x16x32, f32 accum,
// async global->LDS staging (ASYNCcnt) for pure-copy tiles.
// ---------------------------------------------------------------------------

typedef __attribute__((ext_vector_type(16))) __bf16 v16bf;
typedef __attribute__((ext_vector_type(8)))  __bf16 v8bf;
typedef __attribute__((ext_vector_type(8)))  float  v8f;

#define DEV __device__ __forceinline__

// float -> bf16 bits, round-to-nearest-even
DEV unsigned short f2bf(float f) {
    union { float f; unsigned u; } x; x.f = f;
    unsigned r = x.u + 0x7FFFu + ((x.u >> 16) & 1u);
    return (unsigned short)(r >> 16);
}

DEV v16bf cat8(v8bf lo, v8bf hi) {
    return __builtin_shufflevector(lo, hi, 0,1,2,3,4,5,6,7,8,9,10,11,12,13,14,15);
}

// A-matrix fragment (16x32 bf16). Per ISA 7.12.2: lane m = lane&15 holds row m;
// elements 0..7 <-> K = 8*hi + e ; elements 8..15 <-> K = 16 + 8*hi + (e-8).
DEV v16bf load_fragA(const unsigned short* p, int hi) {
    v8bf lo = *(const v8bf*)(p + 8 * hi);
    v8bf h8 = *(const v8bf*)(p + 16 + 8 * hi);
    return cat8(lo, h8);
}

// B-matrix fragment (32x16 bf16). lane n = lane&15 holds column n; element K order matches A.
DEV v16bf load_fragB(const unsigned short* p, int ld, int hi) {
    v16bf r;
#pragma unroll
    for (int s = 0; s < 8; ++s) {
        r[s]     = __builtin_bit_cast(__bf16, p[(8 * hi + s) * ld]);
        r[s + 8] = __builtin_bit_cast(__bf16, p[(16 + 8 * hi + s) * ld]);
    }
    return r;
}

DEV v8f wmma_bf16(v16bf a, v16bf b, v8f c) {
    return __builtin_amdgcn_wmma_f32_16x16x32_bf16(
        /*neg_a=*/false, a, /*neg_b=*/false, b,
        /*c_mod=*/(short)0, c, /*reuse_a=*/false, /*reuse_b=*/false);
}

// ---- async global -> LDS copies (tracked on ASYNCcnt), gfx1250 ------------
// Generic LDS pointer low 32 bits == offset within the wave's LDS allocation.
DEV unsigned lds_off(const void* p) { return (unsigned)(uintptr_t)p; }

DEV void async_copy_b128(unsigned lds, const void* gaddr) {
    asm volatile("global_load_async_to_lds_b128 %0, %1, off"
                 :: "v"(lds), "v"(gaddr) : "memory");
}
DEV void async_copy_b64(unsigned lds, const void* gaddr) {
    asm volatile("global_load_async_to_lds_b64 %0, %1, off"
                 :: "v"(lds), "v"(gaddr) : "memory");
}
DEV void wait_async0() {
    asm volatile("s_wait_asynccnt 0" ::: "memory");
}

// ---------------------------------------------------------------------------
// Problem constants
// ---------------------------------------------------------------------------
constexpr int SEQ  = 2048;
constexpr int DMOD = 2048;
constexpr int HEADS = 16;
constexpr int DH   = 128;
constexpr int KDIM = 2048;                 // reduction dim of both projections
constexpr float LN10K_OVER_DH = 0.0719557841560639f;   // ln(10000)/128
constexpr float SM_SCALE = 0.08838834764831845f;       // 1/sqrt(128)

// ---------------------------------------------------------------------------
// GEMM:  C[M,N] = A[M,K] @ W[N,K]^T (+bias)
//   MODE 0: A = x (fp32), N=6144; epilogue = bias + RoPE, scatter q/k/v bf16
//   MODE 1: A = y (bf16 ws, async-copied to LDS), N=2048; bias + fp32 store.
// Block tile 128x128, K-step 32, 256 threads = 8 waves (4 M x 2 N),
// each wave computes 32x64 = 2x4 WMMA tiles.
// ---------------------------------------------------------------------------
template <int MODE>
__global__ __launch_bounds__(256, 2)
void gemm_kernel(const float* __restrict__ Afp,
                 const unsigned short* __restrict__ Abf,
                 const float* __restrict__ W,
                 const float* __restrict__ bias,
                 float* __restrict__ outF,
                 unsigned short* __restrict__ qb,
                 unsigned short* __restrict__ kb,
                 unsigned short* __restrict__ vb)
{
    __shared__ unsigned short As[128][40];   // [M][K] k-contiguous (+pad)
    __shared__ unsigned short Bs[32][136];   // [K][N] n-contiguous (+pad)

    const int tid  = threadIdx.x;
    const int lane = tid & 31;
    const int wid  = tid >> 5;
    const int wm   = wid & 3;       // wave M slot (32 rows)
    const int wn   = wid >> 2;      // wave N slot (64 cols)
    const int ln   = lane & 15;
    const int hi   = lane >> 4;

    const int m0 = blockIdx.y * 128;
    const int n0 = blockIdx.x * 128;

    const int lr  = tid >> 3;        // 0..31
    const int lc4 = (tid & 7) * 4;   // 0..28

    v8f acc[2][4];
#pragma unroll
    for (int i = 0; i < 2; ++i)
#pragma unroll
        for (int j = 0; j < 4; ++j)
#pragma unroll
            for (int e = 0; e < 8; ++e) acc[i][j][e] = 0.f;

    for (int k0 = 0; k0 < KDIM; k0 += 32) {
        // ---- stage A tile ----
#pragma unroll
        for (int i = 0; i < 4; ++i) {
            int m = lr + i * 32;
            if (MODE == 0) {
                // fp32 -> bf16 conversion must go through VGPRs
                float4 v = *(const float4*)(Afp + (size_t)(m0 + m) * KDIM + k0 + lc4);
                unsigned p0 = (unsigned)f2bf(v.x) | ((unsigned)f2bf(v.y) << 16);
                unsigned p1 = (unsigned)f2bf(v.z) | ((unsigned)f2bf(v.w) << 16);
                *(unsigned*)&As[m][lc4]     = p0;
                *(unsigned*)&As[m][lc4 + 2] = p1;
            } else {
                // pure bf16 copy: async global -> LDS, no VGPR round-trip
                async_copy_b64(lds_off(&As[m][lc4]),
                               Abf + (size_t)(m0 + m) * KDIM + k0 + lc4);
            }
        }
        // ---- stage W tile transposed: Bs[k][n] = bf16(W[n][k]) ----
#pragma unroll
        for (int i = 0; i < 4; ++i) {
            int n = lr + i * 32;
            float4 w = *(const float4*)(W + (size_t)(n0 + n) * KDIM + k0 + lc4);
            Bs[lc4 + 0][n] = f2bf(w.x);
            Bs[lc4 + 1][n] = f2bf(w.y);
            Bs[lc4 + 2][n] = f2bf(w.z);
            Bs[lc4 + 3][n] = f2bf(w.w);
        }
        if (MODE == 1) wait_async0();
        __syncthreads();

        v16bf af[2];
#pragma unroll
        for (int mf = 0; mf < 2; ++mf)
            af[mf] = load_fragA(&As[wm * 32 + mf * 16 + ln][0], hi);
        v16bf bfr[4];
#pragma unroll
        for (int nf = 0; nf < 4; ++nf)
            bfr[nf] = load_fragB(&Bs[0][wn * 64 + nf * 16 + ln], 136, hi);
#pragma unroll
        for (int mf = 0; mf < 2; ++mf)
#pragma unroll
            for (int nf = 0; nf < 4; ++nf)
                acc[mf][nf] = wmma_bf16(af[mf], bfr[nf], acc[mf][nf]);
        __syncthreads();
    }

    // ---- epilogue ----
#pragma unroll
    for (int mf = 0; mf < 2; ++mf) {
#pragma unroll
        for (int nf = 0; nf < 4; ++nf) {
            const int colbase = n0 + wn * 64 + nf * 16;
            const int col = colbase + ln;
            const float bv = bias[col];
            if (MODE == 1) {
#pragma unroll
                for (int r = 0; r < 8; ++r) {
                    int m = m0 + wm * 32 + mf * 16 + r + 8 * hi;
                    outF[(size_t)m * DMOD + col] = acc[mf][nf][r] + bv;
                }
            } else {
                const int which = colbase >> 11;          // 0=q 1=k 2=v (uniform per frag)
                const int d  = col & 2047;
                const int hh = d >> 7;
                const int dd = d & 127;
                unsigned short* dstbase = (which == 0) ? qb : (which == 1) ? kb : vb;
                const float invfreq = __expf(-(float)(dd & ~1) * LN10K_OVER_DH);
#pragma unroll
                for (int r = 0; r < 8; ++r) {
                    int m = m0 + wm * 32 + mf * 16 + r + 8 * hi;
                    int bb   = m >> 11;
                    int spos = m & 2047;
                    float val = acc[mf][nf][r] + bv;
                    if (which < 2) {   // RoPE for q and k; pair partner = adjacent lane
                        float partner = __shfl_xor(val, 1, 32);
                        float ang = (float)spos * invfreq;
                        float sn, cs;
                        __sincosf(ang, &sn, &cs);
                        val = ((dd & 1) == 0) ? (val * cs - partner * sn)
                                              : (partner * sn + val * cs);
                    }
                    size_t dst = (((size_t)bb * HEADS + hh) * SEQ + (size_t)spos) * DH + dd;
                    dstbase[dst] = f2bf(val);
                }
            }
        }
    }
}

// ---------------------------------------------------------------------------
// Flash attention: grid (S/128, H, B), 256 threads = 8 waves.
// Each wave owns 16 query rows; Q fragments live in registers for the whole
// kernel. V tile staged with async global->LDS copies (ASYNCcnt); K tile
// staged transposed through VGPRs (transpose needs VALU). Online softmax over
// 32-key chunks; P converted C-layout -> A-layout through per-wave LDS
// scratch, then O += P.V via WMMA.
// ---------------------------------------------------------------------------
__global__ __launch_bounds__(256, 1)
void fa_kernel(const unsigned short* __restrict__ qb,
               const unsigned short* __restrict__ kb,
               const unsigned short* __restrict__ vb,
               unsigned short* __restrict__ yb)
{
    __shared__ unsigned short KsT[128][72];   // [d][t]  (transposed K tile)
    __shared__ unsigned short Vs[64][136];    // [t][d]
    __shared__ unsigned short Ps[8][16][40];  // per-wave P scratch [m][t]

    const int tid  = threadIdx.x;
    const int lane = tid & 31;
    const int wid  = tid >> 5;
    const int ln   = lane & 15;
    const int hi   = lane >> 4;

    const int qt = blockIdx.x;
    const int h  = blockIdx.y;
    const int b  = blockIdx.z;
    const int qbase = qt * 128;
    const size_t headoff = ((size_t)(b * HEADS + h)) * SEQ * DH;

    // Q fragments: 16 rows x 128 d, 4 chunks of K=32 -> registers, from global
    v16bf qf[4];
    {
        const unsigned short* qp = qb + headoff + (size_t)(qbase + wid * 16 + ln) * DH;
#pragma unroll
        for (int kc = 0; kc < 4; ++kc)
            qf[kc] = load_fragA(qp + kc * 32, hi);
    }

    float Mr[8], Lr[8];
    v8f O[8];
#pragma unroll
    for (int r = 0; r < 8; ++r) { Mr[r] = -1e30f; Lr[r] = 0.f; }
#pragma unroll
    for (int f = 0; f < 8; ++f)
#pragma unroll
        for (int r = 0; r < 8; ++r) O[f][r] = 0.f;

    const int nkt = qt * 2 + 2;               // key tiles of 64 up to causal bound
    for (int kt = 0; kt < nkt; ++kt) {
        const int tbase = kt * 64;
        __syncthreads();
        {   // stage K (transposed, via VGPRs) and V (async copy) tiles
            const int tr = tid >> 2;          // key row 0..63
            const int c0 = (tid & 3) * 32;    // d offset
            const unsigned short* kr = kb + headoff + (size_t)(tbase + tr) * DH + c0;
            const unsigned short* vr = vb + headoff + (size_t)(tbase + tr) * DH + c0;
#pragma unroll
            for (int j = 0; j < 4; ++j) {
                async_copy_b128(lds_off(&Vs[tr][c0 + j * 8]), vr + j * 8);
                uint4 kv = *(const uint4*)(kr + j * 8);
                const unsigned short* ks = (const unsigned short*)&kv;
#pragma unroll
                for (int e = 0; e < 8; ++e) KsT[c0 + j * 8 + e][tr] = ks[e];
            }
            // prefetch next tile into cache while this one is consumed
            if (kt + 1 < nkt) {
                __builtin_prefetch(kr + 64 * DH, 0, 0);
                __builtin_prefetch(vr + 64 * DH, 0, 0);
            }
        }
        wait_async0();
        __syncthreads();

#pragma unroll
        for (int c = 0; c < 2; ++c) {         // two 32-key chunks per tile
            const int t0 = c * 32;
            v8f s0, s1;
#pragma unroll
            for (int e = 0; e < 8; ++e) { s0[e] = 0.f; s1[e] = 0.f; }
#pragma unroll
            for (int kc = 0; kc < 4; ++kc) {  // dh reduction
                v16bf b0 = load_fragB(&KsT[kc * 32][t0 + ln], 72, hi);
                v16bf b1 = load_fragB(&KsT[kc * 32][t0 + 16 + ln], 72, hi);
                s0 = wmma_bf16(qf[kc], b0, s0);
                s1 = wmma_bf16(qf[kc], b1, s1);
            }
            // scale + causal mask
            const int tg0 = tbase + t0 + ln;
            const int tg1 = tg0 + 16;
            float rm[8];
#pragma unroll
            for (int r = 0; r < 8; ++r) {
                const int mrow = qbase + wid * 16 + r + 8 * hi;
                s0[r] = (tg0 <= mrow) ? s0[r] * SM_SCALE : -1e30f;
                s1[r] = (tg1 <= mrow) ? s1[r] * SM_SCALE : -1e30f;
                rm[r] = fmaxf(s0[r], s1[r]);
            }
#pragma unroll
            for (int off = 1; off < 16; off <<= 1)
#pragma unroll
                for (int r = 0; r < 8; ++r)
                    rm[r] = fmaxf(rm[r], __shfl_xor(rm[r], off, 32));
            float alpha[8], psum[8];
#pragma unroll
            for (int r = 0; r < 8; ++r) {
                float nm = fmaxf(Mr[r], rm[r]);
                alpha[r] = __expf(Mr[r] - nm);
                Mr[r] = nm;
                float p0 = __expf(s0[r] - nm);
                float p1 = __expf(s1[r] - nm);
                s0[r] = p0; s1[r] = p1;
                psum[r] = p0 + p1;
            }
#pragma unroll
            for (int off = 1; off < 16; off <<= 1)
#pragma unroll
                for (int r = 0; r < 8; ++r)
                    psum[r] += __shfl_xor(psum[r], off, 32);
#pragma unroll
            for (int r = 0; r < 8; ++r) Lr[r] = Lr[r] * alpha[r] + psum[r];
#pragma unroll
            for (int f = 0; f < 8; ++f)
#pragma unroll
                for (int r = 0; r < 8; ++r) O[f][r] *= alpha[r];

            // P: C-layout -> per-wave LDS scratch -> A-layout fragment
#pragma unroll
            for (int r = 0; r < 8; ++r) {
                Ps[wid][r + 8 * hi][ln]      = f2bf(s0[r]);
                Ps[wid][r + 8 * hi][16 + ln] = f2bf(s1[r]);
            }
            asm volatile("s_wait_dscnt 0" ::: "memory");
            v16bf pf = load_fragA(&Ps[wid][ln][0], hi);

            // O += P (16x32) . V (32x128)
#pragma unroll
            for (int f = 0; f < 8; ++f) {
                v16bf vf = load_fragB(&Vs[t0][f * 16 + ln], 136, hi);
                O[f] = wmma_bf16(pf, vf, O[f]);
            }
        }
    }

    // epilogue: O /= L, write bf16 y[B,S,D]
#pragma unroll
    for (int r = 0; r < 8; ++r) {
        const float inv = 1.f / Lr[r];
        const int sg = qbase + wid * 16 + r + 8 * hi;
#pragma unroll
        for (int f = 0; f < 8; ++f) {
            const int dg = h * DH + f * 16 + ln;
            yb[((size_t)b * SEQ + sg) * DMOD + dg] = f2bf(O[f][r] * inv);
        }
    }
}

// ---------------------------------------------------------------------------
// Host launcher
// ---------------------------------------------------------------------------
extern "C" void kernel_launch(void* const* d_in, const int* in_sizes, int n_in,
                              void* d_out, int out_size, void* d_ws, size_t ws_size,
                              hipStream_t stream)
{
    (void)in_sizes; (void)n_in; (void)out_size; (void)ws_size;
    const float* x     = (const float*)d_in[0];
    const float* qkv_w = (const float*)d_in[1];
    const float* qkv_b = (const float*)d_in[2];
    const float* out_w = (const float*)d_in[3];
    const float* out_b = (const float*)d_in[4];
    float* out = (float*)d_out;

    unsigned short* ws = (unsigned short*)d_ws;
    const size_t TSZ = (size_t)4 * HEADS * SEQ * DH;   // 16,777,216 elems (32 MB bf16)
    unsigned short* qbuf = ws;
    unsigned short* kbuf = ws + TSZ;
    unsigned short* vbuf = ws + 2 * TSZ;
    unsigned short* ybuf = ws + 3 * TSZ;

    // 1) qkv = x @ qkv_w.T + b, RoPE(q,k), scatter to [B,H,S,dh] bf16
    gemm_kernel<0><<<dim3(48, 64), 256, 0, stream>>>(
        x, nullptr, qkv_w, qkv_b, nullptr, qbuf, kbuf, vbuf);
    // 2) causal flash attention -> y [B,S,D] bf16
    fa_kernel<<<dim3(16, 16, 4), 256, 0, stream>>>(qbuf, kbuf, vbuf, ybuf);
    // 3) out = y @ out_w.T + out_b  (fp32)
    gemm_kernel<1><<<dim3(16, 64), 256, 0, stream>>>(
        nullptr, ybuf, out_w, out_b, out, nullptr, nullptr, nullptr);
}
